// GraphConvNet_31980326486806
// MI455X (gfx1250) — matrix operand order repaired and checked by reading, not compiled, not run on hardware
//
#include <hip/hip_runtime.h>

typedef _Float16 v8h  __attribute__((ext_vector_type(8)));
typedef _Float16 v16h __attribute__((ext_vector_type(16)));
typedef float    v8f  __attribute__((ext_vector_type(8)));

#define BATCH 8
#define NV 2048
#define BM 64
#define BN 64
#define BKK 64          // K elements staged per iteration (2 wmma k-steps of 32)
#define LPAD 72         // LDS row stride in halves (144B: 16B-aligned, conflict-free)

// Async global->LDS (CDNA5): use if the toolchain exposes the builtin.
#if defined(__has_builtin)
# if __has_builtin(__builtin_amdgcn_global_load_async_to_lds_b128)
#  define USE_ASYNC_LDS 1
# endif
#endif
#ifndef USE_ASYNC_LDS
# define USE_ASYNC_LDS 0
#endif

// Builtin parameter types (from compiler diagnostic): global AS(1) int4*, LDS AS(3) int4*.
typedef int v4i_sz __attribute__((__vector_size__(16)));
typedef __attribute__((address_space(1))) v4i_sz* g4p;
typedef __attribute__((address_space(3))) v4i_sz* l4p;

__device__ __forceinline__ void wait_async_lds()
{
#if USE_ASYNC_LDS
# if defined(__has_builtin) && __has_builtin(__builtin_amdgcn_s_wait_asynccnt)
    __builtin_amdgcn_s_wait_asynccnt(0);
# else
    asm volatile("s_wait_asynccnt 0" ::: "memory");
# endif
#endif
}

// ---------------------------------------------------------------------------
// f16 WMMA GEMM:  D[M,N] = alpha * A[M,K] @ BT[N,K]^T  [- XprevT^T] [+ bias] [relu]
//   A : row-major [M,K]  (f16; or f32 converted during staging if AF32)
//   BT: row-major [N,K]  f16  (B column-major -> direct LDS copy, no transpose)
// Outputs (any subset): Cf (f32 [M,N]), Cn (f16 [M,N]), Ct (f16 transposed [N,M]).
// XprevT (f16 [N,M]) subtracted if non-null. Batched via blockIdx.z strides.
// Block = 128 threads (4 waves); block tile 64x64; wave tile 32x32 (2x2 WMMA).
// ---------------------------------------------------------------------------
template <int AF32>
__global__ void __launch_bounds__(128) wmma_gemm_f16(
    const void* __restrict__ Aptr, long long sA, int lda,
    const _Float16* __restrict__ BT, long long sBT, int ldbt,
    float* __restrict__ Cf, long long sCf, int ldcf,
    _Float16* __restrict__ Cn, long long sCn, int ldcn,
    _Float16* __restrict__ Ct, long long sCt, int ldct,
    const _Float16* __restrict__ XprevT, long long sX, int ldx,
    const float* __restrict__ bias,
    int Kg, float alpha, int do_relu)
{
    const int bz = blockIdx.z;
    const float*    Af = (const float*)Aptr + (long long)bz * sA;
    const _Float16* Ah = (const _Float16*)Aptr + (long long)bz * sA;
    BT += (long long)bz * sBT;
    if (Cf) Cf += (long long)bz * sCf;
    if (Cn) Cn += (long long)bz * sCn;
    if (Ct) Ct += (long long)bz * sCt;
    if (XprevT) XprevT += (long long)bz * sX;

    const int tileM = blockIdx.y * BM;
    const int tileN = blockIdx.x * BN;

    __shared__ _Float16 As[BM][LPAD];
    __shared__ _Float16 Bs[BN][LPAD];

    const int t    = threadIdx.x;
    const int wave = t >> 5;
    const int lane = t & 31;
    const int hl   = lane >> 4;
    const int l16  = lane & 15;
    const int wm   = (wave >> 1) * 32;
    const int wn   = (wave & 1) * 32;

    v8f acc[2][2] = {};

    for (int kk = 0; kk < Kg; kk += BKK) {
        __syncthreads();
        // Stage A (64x64 halves) and BT (64x64 halves); 512 vec8 each, 128 thr x 4
#pragma unroll
        for (int i = 0; i < 4; ++i) {
            int idx = t + i * 128;
            int r   = idx >> 3;
            int c   = (idx & 7) * 8;
            if (AF32) {
                const float* src = Af + (long long)(tileM + r) * lda + kk + c;
                float4 f0 = *(const float4*)(src);
                float4 f1 = *(const float4*)(src + 4);
                v8h h;
                h[0] = (_Float16)f0.x; h[1] = (_Float16)f0.y;
                h[2] = (_Float16)f0.z; h[3] = (_Float16)f0.w;
                h[4] = (_Float16)f1.x; h[5] = (_Float16)f1.y;
                h[6] = (_Float16)f1.z; h[7] = (_Float16)f1.w;
                *(v8h*)&As[r][c] = h;
            } else {
#if USE_ASYNC_LDS
                __builtin_amdgcn_global_load_async_to_lds_b128(
                    (g4p)(Ah + (long long)(tileM + r) * lda + kk + c),
                    (l4p)&As[r][c], 0, 0);
#else
                *(v8h*)&As[r][c] =
                    *(const v8h*)(Ah + (long long)(tileM + r) * lda + kk + c);
#endif
            }
#if USE_ASYNC_LDS
            __builtin_amdgcn_global_load_async_to_lds_b128(
                (g4p)(BT + (long long)(tileN + r) * ldbt + kk + c),
                (l4p)&Bs[r][c], 0, 0);
#else
            *(v8h*)&Bs[r][c] =
                *(const v8h*)(BT + (long long)(tileN + r) * ldbt + kk + c);
#endif
        }
        wait_async_lds();
        __syncthreads();

#pragma unroll
        for (int ks = 0; ks < BKK; ks += 32) {
            // A fragment: lane m, halves K = ks + 8*hl + [0..7] and ks + 16 + 8*hl + [0..7]
            v16h afrag[2];
#pragma unroll
            for (int mi = 0; mi < 2; ++mi) {
                int m = wm + mi * 16 + l16;
                ((v8h*)&afrag[mi])[0] = *(const v8h*)&As[m][ks + 8 * hl];
                ((v8h*)&afrag[mi])[1] = *(const v8h*)&As[m][ks + 16 + 8 * hl];
            }
            // B fragment: lane n, halves K = ks + 8*hl + [0..7] and ks + 16 + 8*hl + [0..7]
            v16h bfrag[2];
#pragma unroll
            for (int nj = 0; nj < 2; ++nj) {
                int n = wn + nj * 16 + l16;
                ((v8h*)&bfrag[nj])[0] = *(const v8h*)&Bs[n][ks + 8 * hl];
                ((v8h*)&bfrag[nj])[1] = *(const v8h*)&Bs[n][ks + 16 + 8 * hl];
            }
#pragma unroll
            for (int mi = 0; mi < 2; ++mi)
#pragma unroll
                for (int nj = 0; nj < 2; ++nj)
                    acc[mi][nj] = __builtin_amdgcn_wmma_f32_16x16x32_f16(
                        false, afrag[mi], false, bfrag[nj],
                        (short)0, acc[mi][nj], false, false);
        }
    }

    // Epilogue: lanes 0-15 -> M = r, lanes 16-31 -> M = 8 + r; N = l16
#pragma unroll
    for (int mi = 0; mi < 2; ++mi) {
#pragma unroll
        for (int nj = 0; nj < 2; ++nj) {
            int n = tileN + wn + nj * 16 + l16;
            float bv = bias ? bias[n] : 0.0f;
#pragma unroll
            for (int r = 0; r < 8; ++r) {
                int m = tileM + wm + mi * 16 + hl * 8 + r;
                float v = alpha * acc[mi][nj][r] + bv;
                if (XprevT) v -= (float)XprevT[(long long)n * ldx + m];
                if (do_relu) v = fmaxf(v, 0.0f);
                if (Cf) Cf[(long long)m * ldcf + n] = v;
                if (Cn) Cn[(long long)m * ldcn + n] = (_Float16)v;
                if (Ct) Ct[(long long)n * ldct + m] = (_Float16)v;
            }
        }
    }
}

// ---------------------------------------------------------------------------
// Xk1 init (f16): Xk1[b*V+v][0:3] = x[b][:][v]; cols 3..63 zero (K pad 18->64).
// ---------------------------------------------------------------------------
__global__ void init_xk1(const float* __restrict__ x, _Float16* __restrict__ Xk1)
{
    int i = blockIdx.x * blockDim.x + threadIdx.x;
    if (i >= BATCH * NV) return;
    int b = i >> 11, v = i & (NV - 1);
    _Float16* row = Xk1 + (long long)i * 64;
#pragma unroll
    for (int c = 0; c < 64; ++c) row[c] = (_Float16)0.0f;
    row[0] = (_Float16)x[((long long)b * 3 + 0) * NV + v];
    row[1] = (_Float16)x[((long long)b * 3 + 1) * NV + v];
    row[2] = (_Float16)x[((long long)b * 3 + 2) * NV + v];
}

// ---------------------------------------------------------------------------
// Conv1 Chebyshev step (Fin=3): wave per (b,v); L read in f32 for accuracy.
// ---------------------------------------------------------------------------
__global__ void cheb1_step(const float* __restrict__ L, _Float16* __restrict__ Xk1,
                           int kidx, float alpha, int hasSub)
{
    int gid  = blockIdx.x * blockDim.x + threadIdx.x;
    int wid  = gid >> 5;
    int lane = gid & 31;
    if (wid >= BATCH * NV) return;
    int b = wid >> 11, v = wid & (NV - 1);
    const float* Lrow = L + ((long long)b * NV + v) * NV;
    const int cprev = (kidx - 1) * 3;
    float s0 = 0.f, s1 = 0.f, s2 = 0.f;
    for (int u = lane; u < NV; u += 32) {
        float a = Lrow[u];
        const _Float16* xr = Xk1 + ((long long)b * NV + u) * 64 + cprev;
        s0 += a * (float)xr[0];
        s1 += a * (float)xr[1];
        s2 += a * (float)xr[2];
    }
#pragma unroll
    for (int off = 16; off > 0; off >>= 1) {
        s0 += __shfl_xor(s0, off, 32);
        s1 += __shfl_xor(s1, off, 32);
        s2 += __shfl_xor(s2, off, 32);
    }
    if (lane == 0) {
        _Float16* o = Xk1 + (long long)wid * 64 + kidx * 3;
        const _Float16* sub = Xk1 + (long long)wid * 64 + (kidx - 2) * 3;
        o[0] = (_Float16)(alpha * s0 - (hasSub ? (float)sub[0] : 0.f));
        o[1] = (_Float16)(alpha * s1 - (hasSub ? (float)sub[1] : 0.f));
        o[2] = (_Float16)(alpha * s2 - (hasSub ? (float)sub[2] : 0.f));
    }
}

// ---------------------------------------------------------------------------
// Weight prep (f16).  W1c[64][64]: col c = k*3+fin (c>=18 zero), src col fin*6+k.
//                     W2c[128][320]: col c = k*64+fin, src col fin*5+k.
// ---------------------------------------------------------------------------
__global__ void prep_w1(const float* __restrict__ w, _Float16* __restrict__ out)
{
    int i = blockIdx.x * blockDim.x + threadIdx.x;
    if (i >= 64 * 64) return;
    int o = i >> 6, c = i & 63;
    float v = 0.0f;
    if (c < 18) { int k = c / 3, fin = c % 3; v = w[o * 18 + fin * 6 + k]; }
    out[i] = (_Float16)v;
}

__global__ void prep_w2(const float* __restrict__ w, _Float16* __restrict__ out)
{
    int i = blockIdx.x * blockDim.x + threadIdx.x;
    if (i >= 128 * 320) return;
    int o = i / 320, c = i % 320;
    int k = c >> 6, fin = c & 63;
    out[i] = (_Float16)w[o * 320 + fin * 5 + k];
}

__global__ void f32_to_f16(const float* __restrict__ src, _Float16* __restrict__ dst, int n)
{
    int i = blockIdx.x * blockDim.x + threadIdx.x;
    if (i < n) dst[i] = (_Float16)src[i];
}

// ---------------------------------------------------------------------------
extern "C" void kernel_launch(void* const* d_in, const int* in_sizes, int n_in,
                              void* d_out, int out_size, void* d_ws, size_t ws_size,
                              hipStream_t stream)
{
    const float* x     = (const float*)d_in[0];  // [8,3,2048]
    const float* L     = (const float*)d_in[1];  // [8,2048,2048]
    const float* cl1_w = (const float*)d_in[2];  // [64,18]
    const float* cl1_b = (const float*)d_in[3];
    const float* cl2_w = (const float*)d_in[4];  // [128,320]
    const float* cl2_b = (const float*)d_in[5];
    const float* fc1_w = (const float*)d_in[6];  // [512,128]
    const float* fc1_b = (const float*)d_in[7];
    const float* fc2_w = (const float*)d_in[8];  // [256,512]
    const float* fc2_b = (const float*)d_in[9];
    float* out = (float*)d_out;                  // [8,2048,256]
    (void)in_sizes; (void)n_in; (void)out_size; (void)ws_size;

    // Workspace (halves)
    _Float16* ws = (_Float16*)d_ws;
    size_t off = 0;
    _Float16* Xk1 = ws + off; off += (size_t)BATCH * NV * 64;    // conv1 concat, K pad 64
    _Float16* W1c = ws + off; off += 64 * 64;
    _Float16* W2c = ws + off; off += 128 * 320;
    _Float16* F1h = ws + off; off += 512 * 128;
    _Float16* F2h = ws + off; off += 256 * 512;
    _Float16* Tt  = ws + off; off += (size_t)BATCH * 320 * NV;   // conv2 concat, transposed [320][V]
    _Float16* Tn  = ws + off; off += (size_t)BATCH * NV * 320;   // conv2 concat, normal [V][320]
    _Float16* Y2h = ws + off; off += (size_t)BATCH * NV * 128;
    _Float16* Y3h = ws + off; off += (size_t)BATCH * NV * 512;

    const int M = BATCH * NV;                 // 16384
    const long long sL = (long long)NV * NV;  // L batch stride
    const long long sT = (long long)320 * NV; // Tt / Tn batch stride (same count)

    // ---- staging + weight prep ------------------------------------------
    init_xk1<<<(BATCH * NV + 255) / 256, 256, 0, stream>>>(x, Xk1);
    prep_w1<<<(64 * 64 + 255) / 256, 256, 0, stream>>>(cl1_w, W1c);
    prep_w2<<<(128 * 320 + 255) / 256, 256, 0, stream>>>(cl2_w, W2c);
    f32_to_f16<<<(512 * 128 + 255) / 256, 256, 0, stream>>>(fc1_w, F1h, 512 * 128);
    f32_to_f16<<<(256 * 512 + 255) / 256, 256, 0, stream>>>(fc2_w, F2h, 256 * 512);

    // ---- conv1 Chebyshev recurrence (Fin=3) ------------------------------
    for (int k = 1; k < 6; ++k)
        cheb1_step<<<(BATCH * NV * 32 + 255) / 256, 256, 0, stream>>>(
            L, Xk1, k, (k == 1) ? 1.0f : 2.0f, k >= 2);

    // ---- conv1 weight apply -> T0 (dual store: Tt slice 0 + Tn cols 0:64)
    {
        dim3 grid(1, NV / BM, BATCH);
        wmma_gemm_f16<0><<<grid, 128, 0, stream>>>(
            Xk1, (long long)NV * 64, 64,
            W1c, 0, 64,
            nullptr, 0, 0,
            Tn, sT, 320,
            Tt, sT, NV,
            nullptr, 0, 0,
            cl1_b, 64, 1.0f, 1);
    }

    // ---- conv2 Chebyshev recurrence: T_k = a*L@T_{k-1} - T_{k-2} ---------
    {
        dim3 grid(1, NV / BM, BATCH);
        for (int k = 1; k < 5; ++k) {
            const _Float16* xprev = (k >= 2) ? (Tt + (long long)(k - 2) * 64 * NV) : nullptr;
            wmma_gemm_f16<1><<<grid, 128, 0, stream>>>(
                L, sL, NV,                                 // A = L (f32, converted in staging)
                Tt + (long long)(k - 1) * 64 * NV, sT, NV, // BT = t_{k-1}^T
                nullptr, 0, 0,
                Tn + k * 64, sT, 320,                      // normal store
                Tt + (long long)k * 64 * NV, sT, NV,       // transposed store
                xprev, sT, NV,
                nullptr, NV, (k == 1) ? 1.0f : 2.0f, 0);
        }
    }

    // ---- conv2 weight apply: Y2 = relu(Tn @ W2c^T + b2) ------------------
    {
        dim3 grid(128 / BN, M / BM, 1);
        wmma_gemm_f16<0><<<grid, 128, 0, stream>>>(
            Tn, 0, 320,  W2c, 0, 320,
            nullptr, 0, 0,  Y2h, 0, 128,  nullptr, 0, 0,
            nullptr, 0, 0,  cl2_b, 320, 1.0f, 1);
    }

    // ---- FC1: Y3 = relu(Y2 @ fc1_w^T + b) --------------------------------
    {
        dim3 grid(512 / BN, M / BM, 1);
        wmma_gemm_f16<0><<<grid, 128, 0, stream>>>(
            Y2h, 0, 128,  F1h, 0, 128,
            nullptr, 0, 0,  Y3h, 0, 512,  nullptr, 0, 0,
            nullptr, 0, 0,  fc1_b, 128, 1.0f, 1);
    }

    // ---- FC2: out = relu(Y3 @ fc2_w^T + b)  (f32 store to d_out) ---------
    {
        dim3 grid(256 / BN, M / BM, 1);
        wmma_gemm_f16<0><<<grid, 128, 0, stream>>>(
            Y3h, 0, 512,  F2h, 0, 512,
            out, 0, 256,  nullptr, 0, 0,  nullptr, 0, 0,
            nullptr, 0, 0,  fc2_b, 512, 1.0f, 1);
    }
}
